// EdgePowersetLayer_53575422050603
// MI455X (gfx1250) — matrix operand order repaired and checked by reading, not compiled, not run on hardware
//
#include <hip/hip_runtime.h>
#include <hip/hip_bf16.h>

// ---------------------------------------------------------------------------
// EdgePowersetLayer, algebraically reduced:
//   out[b,u] = sum_j x[b,j] * c[j],  c[j] = sum_{subsets s containing j} softmax(w)[s]
// Stage 1: fold 65536 weights -> 16 coefficients (softmax + masked accumulate)
// Stage 2: WMMA f16 hi/lo split dot (x[1024,16] . c[16]) broadcast to [1024,64]
// ---------------------------------------------------------------------------

typedef __attribute__((ext_vector_type(16))) _Float16 v16h;
typedef __attribute__((ext_vector_type(8)))  float    v8f;

#define IN_FEATURES 16
#define OUT_FEATURES 64
#define BATCH 1024
#define NUM_SUBSETS 65536

// Pascal triangle C(n,k), n,k in [0,16], zero-padded where k > n.
__constant__ unsigned BINOM[17][17] = {
  {1,0,0,0,0,0,0,0,0,0,0,0,0,0,0,0,0},
  {1,1,0,0,0,0,0,0,0,0,0,0,0,0,0,0,0},
  {1,2,1,0,0,0,0,0,0,0,0,0,0,0,0,0,0},
  {1,3,3,1,0,0,0,0,0,0,0,0,0,0,0,0,0},
  {1,4,6,4,1,0,0,0,0,0,0,0,0,0,0,0,0},
  {1,5,10,10,5,1,0,0,0,0,0,0,0,0,0,0,0},
  {1,6,15,20,15,6,1,0,0,0,0,0,0,0,0,0,0},
  {1,7,21,35,35,21,7,1,0,0,0,0,0,0,0,0,0},
  {1,8,28,56,70,56,28,8,1,0,0,0,0,0,0,0,0},
  {1,9,36,84,126,126,84,36,9,1,0,0,0,0,0,0,0},
  {1,10,45,120,210,252,210,120,45,10,1,0,0,0,0,0,0},
  {1,11,55,165,330,462,462,330,165,55,11,1,0,0,0,0,0},
  {1,12,66,220,495,792,924,792,495,220,66,12,1,0,0,0,0},
  {1,13,78,286,715,1287,1716,1716,1287,715,286,78,13,1,0,0,0},
  {1,14,91,364,1001,2002,3003,3432,3003,2002,1001,364,91,14,1,0,0},
  {1,15,105,455,1365,3003,5005,6435,6435,5005,3003,1365,455,105,15,1,0},
  {1,16,120,560,1820,4368,8008,11440,12870,11440,8008,4368,1820,560,120,16,1}
};

// Map powerset-order index s (itertools: subsets by size r, combos in lex
// order) to its 16-bit membership mask via combinatorial unranking.
__device__ __forceinline__ unsigned powerset_mask(unsigned s) {
  unsigned rem = s;
  int r = 0;
  // find subset size r: subtract C(16,r) block sizes
  while (r < 16 && rem >= BINOM[16][r]) { rem -= BINOM[16][r]; ++r; }
  // lexicographic unranking of the r-combination of {0..15}
  unsigned mask = 0u;
  int kk = r;
  int e = 0;
  while (kk > 0) {
    unsigned cnt = BINOM[15 - e][kk - 1];   // combos whose next element is e
    if (rem < cnt) { mask |= (1u << e); --kk; ++e; }
    else           { rem -= cnt; ++e; }
  }
  return mask;
}

// ---------------------------------------------------------------------------
// Kernel 1: one block of 1024 threads. 64 weights per thread (coalesced,
// stride 1024). Computes softmax-normalized per-feature coefficients c[16].
// ---------------------------------------------------------------------------
__global__ __launch_bounds__(1024)
void powerset_fold_kernel(const float* __restrict__ w, float* __restrict__ c_out) {
  __shared__ float sred[1024];
  __shared__ float sacc[17];               // c[0..15], Z at [16]

  const int tid = threadIdx.x;

  // --- max reduce (numerical stability, matches jax softmax) ---
  float m = -3.402823466e38f;
  #pragma unroll 4
  for (int i = 0; i < 64; ++i) {
    m = fmaxf(m, w[tid + i * 1024]);
  }
  sred[tid] = m;
  __syncthreads();
  for (int off = 512; off > 0; off >>= 1) {
    if (tid < off) sred[tid] = fmaxf(sred[tid], sred[tid + off]);
    __syncthreads();
  }
  const float wmax = sred[0];

  if (tid < 17) sacc[tid] = 0.0f;
  __syncthreads();

  // --- fold exp(w - max) into per-feature sums via unranked masks ---
  float cl[16];
  #pragma unroll
  for (int j = 0; j < 16; ++j) cl[j] = 0.0f;
  float zl = 0.0f;

  for (int i = 0; i < 64; ++i) {
    const unsigned s = (unsigned)tid + (unsigned)i * 1024u;
    if (i < 63) __builtin_prefetch(&w[s + 1024], 0, 1);  // global_prefetch_b8
    const float e = expf(w[s] - wmax);
    zl += e;
    const unsigned mask = powerset_mask(s);
    #pragma unroll
    for (int j = 0; j < 16; ++j) {
      if (mask & (1u << j)) cl[j] += e;
    }
  }

  #pragma unroll
  for (int j = 0; j < 16; ++j) atomicAdd(&sacc[j], cl[j]);   // ds_add_f32
  atomicAdd(&sacc[16], zl);
  __syncthreads();

  if (tid < 16) c_out[tid] = sacc[tid] / sacc[16];
}

// ---------------------------------------------------------------------------
// Kernel 2: 64 blocks x 1 wave (32 lanes). Each wave computes a 16-row tile:
//   D[m][n] = sum_k x[row_m, k] * c[k]   via two chained
//   v_wmma_f32_16x16x32_f16 ops with hi/lo f16 operand splitting (~f32 acc).
// D's columns are identical, so each lane fans its row scalars out to 4
// contiguous output columns with b128 stores.
// ---------------------------------------------------------------------------
__global__ __launch_bounds__(32)
void powerset_out_wmma_kernel(const float* __restrict__ x,
                              const float* __restrict__ c,
                              float* __restrict__ out) {
  const int lane = threadIdx.x;            // 0..31
  const int tile = blockIdx.x;             // 0..63 (16 batch rows each)
  const int half = lane >> 4;              // lane group
  const int l16  = lane & 15;

  // B fragments: every column of B is c (period-16 in K), so each lane's
  // fragment is c[8*half + (k&7)] (B layout matches A: lanes 0-15 hold
  // K=0..7,16..23, lanes 16-31 hold K=8..15,24..31; B is K-periodic with period 16).
  v16h b_hi, b_lo;
  #pragma unroll
  for (int k = 0; k < 16; ++k) {
    const float cv = c[half * 8 + (k & 7)];
    const _Float16 hi = (_Float16)cv;
    const _Float16 lo = (_Float16)(cv - (float)hi);
    b_hi[k] = hi;
    b_lo[k] = lo;
  }

  // A fragment: row m = l16 of the tile. 16-bit A 16x32 layout:
  //   lanes 0-15 : halves 0..7 = K 0..7,  halves 8..15 = K 16..23
  //   lanes 16-31: halves 0..7 = K 8..15, halves 8..15 = K 24..31
  // We place x_hi at K=0..15 and x_lo at K=16..31.
  const int row   = tile * 16 + l16;
  const int kbase = half * 8;
  const float4* xr = (const float4*)(x + row * IN_FEATURES + kbase);
  const float4 p0 = xr[0];
  const float4 p1 = xr[1];
  const float xv[8] = {p0.x, p0.y, p0.z, p0.w, p1.x, p1.y, p1.z, p1.w};

  v16h a;
  #pragma unroll
  for (int k = 0; k < 8; ++k) {
    const _Float16 hi = (_Float16)xv[k];
    const _Float16 lo = (_Float16)(xv[k] - (float)hi);
    a[k]     = hi;    // K = kbase + k         (x_hi)
    a[8 + k] = lo;    // K = 16 + kbase + k    (x_lo)
  }

  v8f d = {};
  // D  = x_hi*c_hi + x_lo*c_hi
  d = __builtin_amdgcn_wmma_f32_16x16x32_f16(false, a, false, b_hi,
                                             (short)0, d, false, false);
  // D += x_hi*c_lo + x_lo*c_lo  ->  (x_hi+x_lo)(c_hi+c_lo) ~= x*c
  d = __builtin_amdgcn_wmma_f32_16x16x32_f16(false, a, false, b_lo,
                                             (short)0, d, false, false);

  // D layout (32-bit 16x16 C/D): lane group 0 covers M=0..7 (vgpr v -> M=v),
  // lane group 1 covers M=8..15. All 16 columns of D are identical (result is
  // a per-row scalar), so lane l16 broadcasts its row scalars to the 4
  // contiguous output columns [4*l16, 4*l16+3] with a single b128 store per
  // row: lanes 0..15 jointly cover all 64 columns, fully coalesced.
  #pragma unroll
  for (int v = 0; v < 8; ++v) {
    const int m = half * 8 + v;
    const float val = d[v];
    const float4 vec = {val, val, val, val};
    float4* o = (float4*)(out + (size_t)(tile * 16 + m) * OUT_FEATURES + 4 * l16);
    *o = vec;   // global_store_b128
  }
}

// ---------------------------------------------------------------------------
extern "C" void kernel_launch(void* const* d_in, const int* in_sizes, int n_in,
                              void* d_out, int out_size, void* d_ws, size_t ws_size,
                              hipStream_t stream) {
  const float* x = (const float*)d_in[0];        // [1024, 16] f32
  const float* w = (const float*)d_in[1];        // [65536]    f32
  float* out  = (float*)d_out;                   // [1024, 64] f32
  float* c_ws = (float*)d_ws;                    // 16 floats of scratch

  powerset_fold_kernel<<<1, 1024, 0, stream>>>(w, c_ws);
  powerset_out_wmma_kernel<<<BATCH / 16, 32, 0, stream>>>(x, c_ws, out);
}
